// ScHetGNN_24154896072768
// MI455X (gfx1250) — matrix operand order repaired and checked by reading, not compiled, not run on hardware
//
#include <hip/hip_runtime.h>

// ---------------------------------------------------------------------------
// Types
// ---------------------------------------------------------------------------
typedef __attribute__((ext_vector_type(16))) _Float16 v16h;
typedef __attribute__((ext_vector_type(8)))  _Float16 v8h;
typedef __attribute__((ext_vector_type(8)))  float    v8f;
typedef __attribute__((ext_vector_type(4)))  float    v4f;

// Problem constants (match reference)
constexpr int D  = 512;      // core dim (K of all GEMMs)
constexpr int E  = 256;      // embedding dim
constexpr int H  = 8;        // heads
constexpr int HD = 64;       // head dim
constexpr int S  = 20;       // neighbors
constexpr int B  = 8192;     // nodes

// ---------------------------------------------------------------------------
// Fragment load helpers (layouts per CDNA5 ISA 7.12.2, wave32)
// ---------------------------------------------------------------------------
__device__ __forceinline__ v16h cat_v8h(v8h lo, v8h hi) {
  v16h a;
#pragma unroll
  for (int e = 0; e < 8; ++e) { a[e] = lo[e]; a[8 + e] = hi[e]; }
  return a;
}

// A 16x32 f16 fragment from f32 source, p = row base + kbA (kb = 0 or 8)
// e in [0,7]: K = kb+e ; e in [8,15]: K = kb+16+(e-8)
__device__ __forceinline__ v16h load_a_f32(const float* p) {
  v4f x0 = __builtin_nontemporal_load((const v4f*)(p));
  v4f x1 = __builtin_nontemporal_load((const v4f*)(p + 4));
  v4f x2 = __builtin_nontemporal_load((const v4f*)(p + 16));
  v4f x3 = __builtin_nontemporal_load((const v4f*)(p + 20));
  v16h a;
#pragma unroll
  for (int e = 0; e < 4; ++e) {
    a[e]      = (_Float16)x0[e];
    a[4 + e]  = (_Float16)x1[e];
    a[8 + e]  = (_Float16)x2[e];
    a[12 + e] = (_Float16)x3[e];
  }
  return a;
}

__device__ __forceinline__ v16h load_a_f16(const _Float16* p) {
  v8h lo = __builtin_nontemporal_load((const v8h*)(p));
  v8h hi = __builtin_nontemporal_load((const v8h*)(p + 16));
  return cat_v8h(lo, hi);
}

// B 32x16 f16 fragment, p = W + n*K + kk + kbB (kb = 0 or 8)
// e in [0,7]: K = kb+e ; e in [8,15]: K = kb+16+(e-8)
__device__ __forceinline__ v16h load_b_frag(const _Float16* p) {
  return cat_v8h(*(const v8h*)(p), *(const v8h*)(p + 16));
}

__device__ __forceinline__ v8f wmma_f16(v16h a, v16h b, v8f c) {
  return __builtin_amdgcn_wmma_f32_16x16x32_f16(false, a, false, b, (short)0,
                                                c, false, false);
}

// ---------------------------------------------------------------------------
// f32 -> f16 weight conversion (vectorized by 4)
// ---------------------------------------------------------------------------
__global__ __launch_bounds__(256) void cvt_f32_f16_v4(
    const float* __restrict__ src, _Float16* __restrict__ dst, int n4) {
  int i = blockIdx.x * blockDim.x + threadIdx.x;
  if (i < n4) {
    v4f x = *(const v4f*)(src + 4 * (size_t)i);
    dst[4 * (size_t)i + 0] = (_Float16)x.x;
    dst[4 * (size_t)i + 1] = (_Float16)x.y;
    dst[4 * (size_t)i + 2] = (_Float16)x.z;
    dst[4 * (size_t)i + 3] = (_Float16)x.w;
  }
}

// ---------------------------------------------------------------------------
// Combined weight: Wc[e][d] = sum_i Wl[e][i] * Wo[i][d]   (f32 accum -> f16)
// grid = (D/256, E), block = 256
// ---------------------------------------------------------------------------
__global__ __launch_bounds__(256) void combine_w_kernel(
    const float* __restrict__ Wl, const float* __restrict__ Wo,
    _Float16* __restrict__ Wc) {
  int d = blockIdx.x * 256 + threadIdx.x;
  int e = blockIdx.y;
  float acc = 0.f;
#pragma unroll 8
  for (int i = 0; i < D; ++i)
    acc = fmaf(Wl[(size_t)e * D + i], Wo[(size_t)i * D + d], acc);
  Wc[(size_t)e * D + d] = (_Float16)acc;
}

// bc[e] = bl[e] + sum_i Wl[e][i] * bo[i]     (1 block of 256)
__global__ __launch_bounds__(256) void combine_b_kernel(
    const float* __restrict__ Wl, const float* __restrict__ bo,
    const float* __restrict__ bl, float* __restrict__ bc) {
  int e = threadIdx.x;
  float acc = bl[e];
  for (int i = 0; i < D; ++i) acc = fmaf(Wl[(size_t)e * D + i], bo[i], acc);
  bc[e] = acc;
}

// ---------------------------------------------------------------------------
// WMMA GEMM:  C[M,N] = A[M,K=512] * W[N,K=512]^T + bias[N]   (f32 accumulate)
// One wave computes a 32(M) x 64(N) tile: 8 accumulators, 2 A frags + 4 B
// frags per k-step, each B frag feeds 2 WMMAs. Software-pipelined and fully
// unrolled: all load addresses become base + 24-bit immediate offsets, and
// the double-buffer swap costs zero register moves.
// ---------------------------------------------------------------------------
template <bool A_F16, bool OUT_F16, bool LEAKY>
__global__ __launch_bounds__(256) void gemm_wmma_kernel(
    const void* __restrict__ Aptr, const _Float16* __restrict__ W,
    const float* __restrict__ bias, void* __restrict__ Cptr, int M, int N) {
  constexpr int K = D;
  const int lane   = threadIdx.x & 31;
  const int wid    = (int)((blockIdx.x * blockDim.x + threadIdx.x) >> 5);
  const int nTiles = N >> 6;      // 64 columns per wave
  const int mTile  = wid / nTiles;
  const int nTile  = wid % nTiles;
  if (mTile * 32 >= M) return;    // wave-uniform: EXEC stays all-ones

  const int half = lane >> 4;     // 0 or 1
  const int l15  = lane & 15;
  const int kbA  = half * 8;
  const int kbB  = half * 8;
  const int row0 = mTile * 32 + l15;
  const int row1 = row0 + 16;
  const int ncol = nTile * 64 + l15;

  auto LA = [&](int row, int kk) -> v16h {
    if constexpr (A_F16)
      return load_a_f16((const _Float16*)Aptr + (size_t)row * K + kk + kbA);
    else
      return load_a_f32((const float*)Aptr + (size_t)row * K + kk + kbA);
  };
  auto LB = [&](int sub, int kk) -> v16h {
    return load_b_frag(W + (size_t)(ncol + sub * 16) * K + kk + kbB);
  };

  v8f acc[8] = {v8f{}, v8f{}, v8f{}, v8f{}, v8f{}, v8f{}, v8f{}, v8f{}};

  v16h a0 = LA(row0, 0), a1 = LA(row1, 0);
  v16h b0 = LB(0, 0), b1 = LB(1, 0), b2 = LB(2, 0), b3 = LB(3, 0);

#pragma unroll
  for (int kk = 0; kk < K - 32; kk += 32) {
    const int kn = kk + 32;
    // prefetch next k-step into fresh registers
    v16h a0n = LA(row0, kn), a1n = LA(row1, kn);
    v16h b0n = LB(0, kn), b1n = LB(1, kn), b2n = LB(2, kn), b3n = LB(3, kn);
    // compute on current fragments (loads above already in flight)
    acc[0] = wmma_f16(a0, b0, acc[0]);
    acc[1] = wmma_f16(a1, b0, acc[1]);
    acc[2] = wmma_f16(a0, b1, acc[2]);
    acc[3] = wmma_f16(a1, b1, acc[3]);
    acc[4] = wmma_f16(a0, b2, acc[4]);
    acc[5] = wmma_f16(a1, b2, acc[5]);
    acc[6] = wmma_f16(a0, b3, acc[6]);
    acc[7] = wmma_f16(a1, b3, acc[7]);
    a0 = a0n; a1 = a1n; b0 = b0n; b1 = b1n; b2 = b2n; b3 = b3n;
  }
  // peeled last k-step
  acc[0] = wmma_f16(a0, b0, acc[0]);
  acc[1] = wmma_f16(a1, b0, acc[1]);
  acc[2] = wmma_f16(a0, b1, acc[2]);
  acc[3] = wmma_f16(a1, b1, acc[3]);
  acc[4] = wmma_f16(a0, b2, acc[4]);
  acc[5] = wmma_f16(a1, b2, acc[5]);
  acc[6] = wmma_f16(a0, b3, acc[6]);
  acc[7] = wmma_f16(a1, b3, acc[7]);

  // ---- Epilogue ----
  // C/D layout: lane L, vgpr j -> M = j + 8*half, N = L&15
#pragma unroll
  for (int sub = 0; sub < 4; ++sub) {
    const int n   = ncol + sub * 16;
    const float bs = bias[n];
#pragma unroll
    for (int mt = 0; mt < 2; ++mt) {
      const v8f& a = acc[sub * 2 + mt];
#pragma unroll
      for (int j = 0; j < 8; ++j) {
        const int m = mTile * 32 + mt * 16 + j + half * 8;
        float v = a[j] + bs;
        if constexpr (LEAKY) v = (v > 0.f) ? v : 0.01f * v;
        if constexpr (OUT_F16)
          ((_Float16*)Cptr)[(size_t)m * N + n] = (_Float16)v;
        else
          ((float*)Cptr)[(size_t)m * N + n] = v;
      }
    }
  }
}

// ---------------------------------------------------------------------------
// Attention: one wave per (node b, head h). Lane l owns head dims l and l+32.
//   Q   : [B, 512] f32
//   KV  : [S*B, 1024] f16 (cols 0..511 = K-proj, 512..1023 = V-proj)
//   ctx : [B, 512] f16
// ---------------------------------------------------------------------------
__global__ __launch_bounds__(256) void attn_kernel(
    const float* __restrict__ Q, const _Float16* __restrict__ KV,
    _Float16* __restrict__ ctx) {
  const int lane = threadIdx.x & 31;
  const int wid  = (int)((blockIdx.x * blockDim.x + threadIdx.x) >> 5);
  const int b = wid >> 3;   // node
  const int h = wid & 7;    // head
  const int d0 = h * HD + lane;

  const float q0 = Q[(size_t)b * D + d0];
  const float q1 = Q[(size_t)b * D + d0 + 32];

  float sc[S];
  const float scale = 0.125f;  // 1/sqrt(64)
#pragma unroll
  for (int s = 0; s < S; ++s) {
    const _Float16* kp = KV + ((size_t)(s * B + b)) * (2 * D) + d0;
    float p = q0 * (float)kp[0] + q1 * (float)kp[32];
#pragma unroll
    for (int off = 16; off > 0; off >>= 1) p += __shfl_xor(p, off, 32);
    sc[s] = p * scale;
  }
  float mx = sc[0];
#pragma unroll
  for (int s = 1; s < S; ++s) mx = fmaxf(mx, sc[s]);
  float sum = 0.f;
#pragma unroll
  for (int s = 0; s < S; ++s) { sc[s] = __expf(sc[s] - mx); sum += sc[s]; }
  const float inv = 1.f / sum;

  float a0 = 0.f, a1 = 0.f;
#pragma unroll
  for (int s = 0; s < S; ++s) {
    const _Float16* vp = KV + ((size_t)(s * B + b)) * (2 * D) + D + d0;
    const float w = sc[s] * inv;
    a0 = fmaf(w, (float)vp[0], a0);
    a1 = fmaf(w, (float)vp[32], a1);
  }
  ctx[(size_t)b * D + d0]      = (_Float16)a0;
  ctx[(size_t)b * D + d0 + 32] = (_Float16)a1;
}

// ---------------------------------------------------------------------------
// Launch
// ---------------------------------------------------------------------------
extern "C" void kernel_launch(void* const* d_in, const int* in_sizes, int n_in,
                              void* d_out, int out_size, void* d_ws, size_t ws_size,
                              hipStream_t stream) {
  const float* query = (const float*)d_in[0];  // [1,B,D]
  const float* keys  = (const float*)d_in[1];  // [S,B,D]
  const float* in_w  = (const float*)d_in[2];  // [3D,D]
  const float* in_b  = (const float*)d_in[3];  // [3D]
  const float* out_w = (const float*)d_in[4];  // [D,D]
  const float* out_b = (const float*)d_in[5];  // [D]
  const float* lin_w = (const float*)d_in[6];  // [E,D]
  const float* lin_b = (const float*)d_in[7];  // [E]
  float* out = (float*)d_out;                  // [B,E]

  // Workspace carve-up (all sub-buffers 256B-aligned)
  char* ws = (char*)d_ws;
  _Float16* W_all = (_Float16*)ws; ws += (size_t)(3 * D) * D * 2;     // Wq|Wk|Wv
  _Float16* Wc    = (_Float16*)ws; ws += (size_t)E * D * 2;           // Wl*Wo
  float*    bc    = (float*)ws;    ws += (size_t)E * 4;               // comb bias
  float*    Qb    = (float*)ws;    ws += (size_t)B * D * 4;           // Q (f32)
  _Float16* KV    = (_Float16*)ws; ws += (size_t)S * B * (2 * D) * 2; // K|V f16
  _Float16* ctx   = (_Float16*)ws;                                    // [B,D] f16

  const _Float16* Wq_h  = W_all;                  // rows [0, D)
  const _Float16* Wkv_h = W_all + (size_t)D * D;  // rows [D, 3D)

  // 1) weight conversion
  {
    int n4 = (3 * D * D) / 4;
    cvt_f32_f16_v4<<<(n4 + 255) / 256, 256, 0, stream>>>(in_w, W_all, n4);
  }
  // 2) combined out_proj+lin weight & bias
  combine_w_kernel<<<dim3(D / 256, E), 256, 0, stream>>>(lin_w, out_w, Wc);
  combine_b_kernel<<<1, 256, 0, stream>>>(lin_w, out_b, lin_b, bc);

  // 3) Q projection: [B,D] = query @ Wq^T + bq   (f32 A, f32 out)
  {
    int M = B, N = D;
    int waves = (M / 32) * (N / 64);
    gemm_wmma_kernel<false, false, false><<<waves / 8, 256, 0, stream>>>(
        (const void*)query, Wq_h, in_b, (void*)Qb, M, N);
  }
  // 4) K|V projection: [S*B, 2D] = keys @ [Wk;Wv]^T + [bk;bv]  (f32 A, f16 out)
  {
    int M = S * B, N = 2 * D;
    int waves = (M / 32) * (N / 64);
    gemm_wmma_kernel<false, true, false><<<waves / 8, 256, 0, stream>>>(
        (const void*)keys, Wkv_h, in_b + D, (void*)KV, M, N);
  }
  // 5) softmax attention -> ctx
  {
    int waves = B * H;  // one wave per (node, head)
    attn_kernel<<<waves / 8, 256, 0, stream>>>(Qb, KV, ctx);
  }
  // 6) fused out_proj+lin: out = LeakyReLU(ctx @ Wc^T + bc)   (f16 A, f32 out)
  {
    int M = B, N = E;
    int waves = (M / 32) * (N / 64);
    gemm_wmma_kernel<true, false, true><<<waves / 8, 256, 0, stream>>>(
        (const void*)ctx, Wc, bc, (void*)out, M, N);
  }
}
